// EncoderBlock_27779848470847
// MI455X (gfx1250) — compile-verified, matrix operand order checked by reading, not run on hardware
//
#include <hip/hip_runtime.h>
#include <math.h>

// ---------------------------------------------------------------- constants
#define BSZ   4
#define SEQ   2048
#define DIM   1024
#define NH    16
#define DK    64
#define DFF   4096
#define EPS   1e-6f
#define NEG_INF (-__builtin_inff())

typedef __attribute__((ext_vector_type(16))) __bf16 v16bf;
typedef __attribute__((ext_vector_type(8)))  __bf16 v8bf;
typedef __attribute__((ext_vector_type(8)))  float  v8f;

// 32-bit LDS byte address for async-to-LDS VDST operand
__device__ __forceinline__ uint32_t lds_off(const void* p) {
  return (uint32_t)(uintptr_t)(__attribute__((address_space(3))) const void*)p;
}

// ------------------------------------------------------------ frag loading
// 16-bit A/B fragment (16x32 / 32x16): per ISA 7.12.2, lane groups 0-15 and
// 16-31 hold the same row; K elements 0..7 -> K base khalf*8, elements 8..15
// -> K base 16+khalf*8.  Two 16-byte LDS reads (ds_load_b128).
__device__ __forceinline__ v16bf load_frag(const __bf16* rowptr, int khalf) {
  v8bf lo = *(const v8bf*)(rowptr + khalf * 8);
  v8bf hi = *(const v8bf*)(rowptr + 16 + khalf * 8);
  v16bf r;
#pragma unroll
  for (int i = 0; i < 8; ++i) { r[i] = lo[i]; r[i + 8] = hi[i]; }
  return r;
}

__device__ __forceinline__ v8f wmma_bf16(v16bf a, v16bf b, v8f c) {
  return __builtin_amdgcn_wmma_f32_16x16x32_bf16(false, a, false, b, (short)0,
                                                 c, false, false);
}

// ------------------------------------------------- weight transpose+convert
// in[rows][cols] fp32  ->  out[cols][rows] bf16   (Wt layout: [N][K])
__global__ __launch_bounds__(256) void convert_T(const float* __restrict__ in,
                                                 __bf16* __restrict__ out,
                                                 int rows, int cols) {
  size_t i = (size_t)blockIdx.x * 256 + threadIdx.x;
  size_t n = (size_t)rows * cols;
  if (i < n) {
    size_t r = i / cols, c = i % cols;
    out[c * (size_t)rows + r] = (__bf16)in[i];
  }
}

// ----------------------------------------------------------------- layernorm
// Reference semantics: unbiased std (ddof=1), divide by (std + eps).
__global__ __launch_bounds__(256) void layernorm_bf16(
    const float* __restrict__ x, const float* __restrict__ w,
    const float* __restrict__ b, __bf16* __restrict__ out) {
  int row = blockIdx.x;
  const float* xr = x + (size_t)row * DIM;
  float s = 0.f, ss = 0.f;
  for (int i = threadIdx.x; i < DIM; i += 256) {
    float v = xr[i]; s += v; ss += v * v;
  }
#pragma unroll
  for (int m = 1; m < 32; m <<= 1) {
    s  += __shfl_xor(s, m, 32);
    ss += __shfl_xor(ss, m, 32);
  }
  __shared__ float sb[2][8];
  int wave = threadIdx.x >> 5;
  if ((threadIdx.x & 31) == 0) { sb[0][wave] = s; sb[1][wave] = ss; }
  __syncthreads();
  if (threadIdx.x == 0) {
    float a = 0.f, c = 0.f;
    for (int i = 0; i < 8; ++i) { a += sb[0][i]; c += sb[1][i]; }
    sb[0][0] = a; sb[1][0] = c;
  }
  __syncthreads();
  float mean = sb[0][0] / DIM;
  float var  = (sb[1][0] - (float)DIM * mean * mean) / (float)(DIM - 1);
  float inv  = 1.f / (sqrtf(var) + EPS);
  for (int i = threadIdx.x; i < DIM; i += 256)
    out[(size_t)row * DIM + i] = (__bf16)(w[i] * (xr[i] - mean) * inv + b[i]);
}

// ------------------------------------------------------------------- GEMM
// C[M,N] = A[M,K](bf16) @ Wt[N,K](bf16)^T + bias (+res) (relu?)
// 128x128 block tile, 8 waves -> 32x64 each, K step 32, bf16 WMMA.
// Double-buffered LDS staging via GLOBAL_LOAD_ASYNC_TO_LDS_B128 (ASYNCcnt):
// 2 async instrs per wave per 128x32 tile per matrix -> 4/wave/tile; waiting
// asynccnt<=4 retires exactly the previous tile (async loads are in-order).
#define BM 128
#define BN 128
#define BKG 32
#define LDT 40  // padded LDS stride (elements); 80B rows keep 16B alignment

// stage one 128x32 bf16 tile (rows gRowBase.., cols k0..k0+31) into lbuf
__device__ __forceinline__ void stage_tile_async(const __bf16* __restrict__ G,
                                                 int gRowBase, int K, int k0,
                                                 __bf16* lbuf, int tid) {
#pragma unroll
  for (int c = tid; c < (BM * BKG) / 8; c += 256) {  // 2 iters/thread
    int r = c >> 2, kc = (c & 3) * 8;
    uint32_t l = lds_off(&lbuf[r * LDT + kc]);
    int off = (int)(((size_t)(gRowBase + r) * K + k0 + kc) * sizeof(__bf16));
    asm volatile("global_load_async_to_lds_b128 %0, %1, %2"
                 :: "v"(l), "v"(off), "s"(G) : "memory");
  }
}

template <bool RELU, bool RES, bool OUT_BF>
__global__ __launch_bounds__(256) void gemm_bf16(
    const __bf16* __restrict__ A, const __bf16* __restrict__ Wt,
    const float* __restrict__ bias, const float* res, float* outF,
    __bf16* outB, int M, int N, int K) {
  __shared__ __bf16 As[2][BM * LDT];
  __shared__ __bf16 Bs[2][BN * LDT];
  const int tid = threadIdx.x, lane = tid & 31, wave = tid >> 5;
  const int khalf = lane >> 4, coll = lane & 15;
  const int wm = wave >> 1, wn = wave & 1;       // 4 x 2 wave grid
  const int m0 = wm * 32, n0 = wn * 64;
  const int bm = blockIdx.y * BM, bn = blockIdx.x * BN;

  v8f acc[2][4] = {};
  stage_tile_async(A, bm, K, 0, As[0], tid);
  stage_tile_async(Wt, bn, K, 0, Bs[0], tid);
  int cur = 0;
  for (int k0 = 0; k0 < K; k0 += BKG, cur ^= 1) {
    if (k0 + BKG < K) {
      stage_tile_async(A, bm, K, k0 + BKG, As[cur ^ 1], tid);
      stage_tile_async(Wt, bn, K, k0 + BKG, Bs[cur ^ 1], tid);
      asm volatile("s_wait_asynccnt 4" ::: "memory");  // retire prev tile only
    } else {
      asm volatile("s_wait_asynccnt 0" ::: "memory");
    }
    __syncthreads();  // whole tile visible (every wave waited on its own)

    v16bf af[2], bfr[4];
#pragma unroll
    for (int i = 0; i < 2; ++i)
      af[i] = load_frag(&As[cur][(m0 + i * 16 + coll) * LDT], khalf);
#pragma unroll
    for (int j = 0; j < 4; ++j)
      bfr[j] = load_frag(&Bs[cur][(n0 + j * 16 + coll) * LDT], khalf);
#pragma unroll
    for (int i = 0; i < 2; ++i)
#pragma unroll
      for (int j = 0; j < 4; ++j)
        acc[i][j] = wmma_bf16(af[i], bfr[j], acc[i][j]);
    __syncthreads();  // WAR: next-next tile's async writes reuse this buffer
  }
  // epilogue: C layout row = r + 8*khalf, col = coll (per 16x16 frag)
#pragma unroll
  for (int i = 0; i < 2; ++i)
#pragma unroll
    for (int j = 0; j < 4; ++j) {
      int col = bn + n0 + j * 16 + coll;
      float bv = bias[col];
#pragma unroll
      for (int r = 0; r < 8; ++r) {
        int row = bm + m0 + i * 16 + khalf * 8 + r;
        float v = acc[i][j][r] + bv;
        if constexpr (RELU) v = fmaxf(v, 0.f);
        if constexpr (RES) v += res[(size_t)row * N + col];
        if constexpr (OUT_BF) outB[(size_t)row * N + col] = (__bf16)v;
        else                  outF[(size_t)row * N + col] = v;
      }
    }
}

// ------------------------------------------------------------ flash attention
// One workgroup: 128 q rows of one (b,h); 8 waves x 16 rows.  Key blocks of
// 32 with online softmax; P routed via wave-private LDS to A-frag layout.
#define BQ 128
#define SKV 32
#define QLD 72  // 64 + 8 pad, 144B rows (16B aligned)

__global__ __launch_bounds__(256) void flash_attn(
    const __bf16* __restrict__ qm, const __bf16* __restrict__ km,
    const __bf16* __restrict__ vm, const int* __restrict__ mask,
    __bf16* __restrict__ outb) {
  __shared__ __bf16 Qs[BQ * QLD];
  __shared__ __bf16 Ks[SKV * QLD];
  __shared__ __bf16 Vt[DK * LDT];            // Vt[d][kv]
  __shared__ __bf16 Ps[8 * 16 * LDT];        // per-wave 16x32 P tile

  const int tid = threadIdx.x, lane = tid & 31, wave = tid >> 5;
  const int khalf = lane >> 4, coll = lane & 15;
  const int b = blockIdx.z, h = blockIdx.y, qb = blockIdx.x * BQ;
  const int qr0 = wave * 16;
  const int wbase = wave * 16 * LDT;

  // stage Q block (128 x 64)
  for (int c = tid; c < BQ * DK / 8; c += 256) {
    int r = c >> 3, kc = (c & 7) * 8;
    *(v8bf*)&Qs[r * QLD + kc] =
        *(const v8bf*)&qm[((size_t)(b * SEQ + qb + r)) * DIM + h * DK + kc];
  }
  __syncthreads();

  float m_r[8], l_r[8], alpha[8];
  v8f o[4] = {};
#pragma unroll
  for (int r = 0; r < 8; ++r) { m_r[r] = NEG_INF; l_r[r] = 0.f; }

  for (int kb = 0; kb < SEQ; kb += SKV) {
    // stage K (32x64) and V^T (64x32)
    for (int c = tid; c < SKV * DK / 8; c += 256) {
      int r = c >> 3, kc = (c & 7) * 8;
      *(v8bf*)&Ks[r * QLD + kc] =
          *(const v8bf*)&km[((size_t)(b * SEQ + kb + r)) * DIM + h * DK + kc];
      v8bf tv =
          *(const v8bf*)&vm[((size_t)(b * SEQ + kb + r)) * DIM + h * DK + kc];
#pragma unroll
      for (int i = 0; i < 8; ++i) Vt[(kc + i) * LDT + r] = tv[i];
    }
    __syncthreads();

    // scores: 16 x 32 tile = 2 frags, K-dim = 64 (2 wmma steps)
    v8f sf[2] = {};
#pragma unroll
    for (int d0 = 0; d0 < DK; d0 += 32) {
      v16bf aq = load_frag(&Qs[(qr0 + coll) * QLD + d0], khalf);
#pragma unroll
      for (int j = 0; j < 2; ++j) {
        v16bf bk = load_frag(&Ks[(j * 16 + coll) * QLD + d0], khalf);
        sf[j] = wmma_bf16(aq, bk, sf[j]);
      }
    }

    int mk0 = mask[(size_t)b * SEQ + kb + coll];
    int mk1 = mask[(size_t)b * SEQ + kb + 16 + coll];
#pragma unroll
    for (int r = 0; r < 8; ++r) {
      float s0 = sf[0][r] * 0.125f;  // 1/sqrt(64)
      float s1 = sf[1][r] * 0.125f;
      if (mk0 == 0) s0 = NEG_INF;
      if (mk1 == 0) s1 = NEG_INF;
      float rmax = fmaxf(s0, s1);
#pragma unroll
      for (int m2 = 1; m2 < 16; m2 <<= 1)
        rmax = fmaxf(rmax, __shfl_xor(rmax, m2, 32));
      float mnew = fmaxf(m_r[r], rmax);
      alpha[r] = (m_r[r] == NEG_INF) ? 0.f : expf(m_r[r] - mnew);
      float p0 = (s0 == NEG_INF) ? 0.f : expf(s0 - mnew);
      float p1 = (s1 == NEG_INF) ? 0.f : expf(s1 - mnew);
      float ls = p0 + p1;
#pragma unroll
      for (int m2 = 1; m2 < 16; m2 <<= 1) ls += __shfl_xor(ls, m2, 32);
      l_r[r] = l_r[r] * alpha[r] + ls;
      m_r[r] = mnew;
      // C layout -> LDS (row r+8*khalf), re-read below in A layout
      Ps[wbase + (r + 8 * khalf) * LDT + coll]      = (__bf16)p0;
      Ps[wbase + (r + 8 * khalf) * LDT + 16 + coll] = (__bf16)p1;
    }
#pragma unroll
    for (int j = 0; j < 4; ++j)
#pragma unroll
      for (int r = 0; r < 8; ++r) o[j][r] *= alpha[r];

    asm volatile("s_wait_dscnt 0" ::: "memory");  // wave-private P RAW fence

    v16bf ap = load_frag(&Ps[wbase + coll * LDT], khalf);
#pragma unroll
    for (int j = 0; j < 4; ++j) {
      v16bf bv = load_frag(&Vt[(j * 16 + coll) * LDT], khalf);
      o[j] = wmma_bf16(ap, bv, o[j]);
    }
    __syncthreads();
  }

  // normalize + store bf16 attention output [b, s, h, dk]
#pragma unroll
  for (int j = 0; j < 4; ++j)
#pragma unroll
    for (int r = 0; r < 8; ++r) {
      int row = qb + qr0 + khalf * 8 + r;
      int col = h * DK + j * 16 + coll;
      float val = (l_r[r] > 0.f) ? o[j][r] / l_r[r] : 0.f;
      outb[((size_t)(b * SEQ + row)) * DIM + col] = (__bf16)val;
    }
}

// ---------------------------------------------------------------- launcher
extern "C" void kernel_launch(void* const* d_in, const int* in_sizes, int n_in,
                              void* d_out, int out_size, void* d_ws,
                              size_t ws_size, hipStream_t stream) {
  (void)in_sizes; (void)n_in; (void)out_size; (void)ws_size;
  const float* x    = (const float*)d_in[0];
  const int*   mask = (const int*)d_in[1];
  const float* wq = (const float*)d_in[2];  const float* bq = (const float*)d_in[3];
  const float* wk = (const float*)d_in[4];  const float* bk = (const float*)d_in[5];
  const float* wv = (const float*)d_in[6];  const float* bv = (const float*)d_in[7];
  const float* wo = (const float*)d_in[8];  const float* bo = (const float*)d_in[9];
  const float* w1 = (const float*)d_in[10]; const float* b1 = (const float*)d_in[11];
  const float* w2 = (const float*)d_in[12]; const float* b2 = (const float*)d_in[13];
  const float* ln1w = (const float*)d_in[14]; const float* ln1b = (const float*)d_in[15];
  const float* ln2w = (const float*)d_in[16]; const float* ln2b = (const float*)d_in[17];
  float* out = (float*)d_out;

  char* ws = (char*)d_ws;
  size_t off = 0;
  auto alloc = [&](size_t bytes) -> void* {
    void* p = ws + off;
    off += (bytes + 255) & ~(size_t)255;
    return p;
  };
  const size_t T = (size_t)BSZ * SEQ;                // 8192 rows
  __bf16* wqt = (__bf16*)alloc((size_t)DIM * DIM * 2);
  __bf16* wkt = (__bf16*)alloc((size_t)DIM * DIM * 2);
  __bf16* wvt = (__bf16*)alloc((size_t)DIM * DIM * 2);
  __bf16* wot = (__bf16*)alloc((size_t)DIM * DIM * 2);
  __bf16* w1t = (__bf16*)alloc((size_t)DIM * DFF * 2);
  __bf16* w2t = (__bf16*)alloc((size_t)DIM * DFF * 2);
  __bf16* xn   = (__bf16*)alloc(T * DIM * 2);
  __bf16* qbuf = (__bf16*)alloc(T * DIM * 2);
  __bf16* kbuf = (__bf16*)alloc(T * DIM * 2);
  __bf16* vbuf = (__bf16*)alloc(T * DIM * 2);
  __bf16* abuf = (__bf16*)alloc(T * DIM * 2);
  __bf16* x1n  = (__bf16*)alloc(T * DIM * 2);
  __bf16* hbuf = (__bf16*)alloc(T * DFF * 2);

  dim3 blk(256);
  int nDD = DIM * DIM, nDF = DIM * DFF;
  convert_T<<<(nDD + 255) / 256, blk, 0, stream>>>(wq, wqt, DIM, DIM);
  convert_T<<<(nDD + 255) / 256, blk, 0, stream>>>(wk, wkt, DIM, DIM);
  convert_T<<<(nDD + 255) / 256, blk, 0, stream>>>(wv, wvt, DIM, DIM);
  convert_T<<<(nDD + 255) / 256, blk, 0, stream>>>(wo, wot, DIM, DIM);
  convert_T<<<(nDF + 255) / 256, blk, 0, stream>>>(w1, w1t, DIM, DFF);
  convert_T<<<(nDF + 255) / 256, blk, 0, stream>>>(w2, w2t, DFF, DIM);

  layernorm_bf16<<<(int)T, blk, 0, stream>>>(x, ln1w, ln1b, xn);

  dim3 gD(DIM / BN, (int)T / BM);        // N=1024
  dim3 gF(DFF / BN, (int)T / BM);        // N=4096
  gemm_bf16<false, false, true><<<gD, blk, 0, stream>>>(
      xn, wqt, bq, nullptr, nullptr, qbuf, (int)T, DIM, DIM);
  gemm_bf16<false, false, true><<<gD, blk, 0, stream>>>(
      xn, wkt, bk, nullptr, nullptr, kbuf, (int)T, DIM, DIM);
  gemm_bf16<false, false, true><<<gD, blk, 0, stream>>>(
      xn, wvt, bv, nullptr, nullptr, vbuf, (int)T, DIM, DIM);

  dim3 ga(SEQ / BQ, NH, BSZ);
  flash_attn<<<ga, blk, 0, stream>>>(qbuf, kbuf, vbuf, mask, abuf);

  // x1 = x + attn @ wo + bo   -> d_out (fp32)
  gemm_bf16<false, true, false><<<gD, blk, 0, stream>>>(
      abuf, wot, bo, x, out, nullptr, (int)T, DIM, DIM);

  layernorm_bf16<<<(int)T, blk, 0, stream>>>(out, ln2w, ln2b, x1n);

  gemm_bf16<true, false, true><<<gF, blk, 0, stream>>>(
      x1n, w1t, b1, nullptr, nullptr, hbuf, (int)T, DFF, DIM);

  // out = x1 + h @ w2 + b2   (in-place residual; each element RMW once)
  gemm_bf16<false, true, false><<<gD, blk, 0, stream>>>(
      hbuf, w2t, b2, out, out, nullptr, (int)T, DIM, DFF);
}
